// InputSENet_790273983045
// MI455X (gfx1250) — compile-verified, matrix operand order checked by reading, not run on hardware
//
#include <hip/hip_runtime.h>
#include <hip/hip_bf16.h>
#include <math.h>

// InputSENet fused kernel for MI455X (gfx1250, wave32).
// One workgroup (256 threads = 8 waves) per 16-row tile of x[16384, 4096].
// Phase A: deterministic segment sums -> field means (f16 in LDS)
// Phase B: SE MLP via v_wmma_f32_16x16x32_f16 (16x128 @ 128x64, 16x64 @ 64x128)
// Phase C: re-read x (L2-hot) and scale by sigmoid gates, non-temporal stores.

typedef _Float16 v16h __attribute__((ext_vector_type(16)));
typedef _Float16 v8h  __attribute__((ext_vector_type(8)));
typedef float    v8f  __attribute__((ext_vector_type(8)));
typedef float    v4f  __attribute__((ext_vector_type(4)));

#define NF   128   // fields
#define NR   64    // hidden (reduction) size
#define DTOT 4096  // features per row
#define ROWS 16    // rows per block (matches WMMA M)
#define NT   256   // threads per block

// chunk t (16 contiguous floats) -> field id. Chunks never straddle fields:
// t<64: len-16 fields (1 chunk each); t in [64,128): len-32 (2 chunks);
// t in [128,256): len-64 (4 chunks).
__device__ __forceinline__ int field_of_chunk(int t) {
  return (t < 64) ? t
       : (t < 128) ? (64 + ((t - 64) >> 1))
                   : (96 + ((t - 128) >> 2));
}

// Build a 16x16x32 f16 WMMA operand (A or B) from a row-major f16 LDS matrix.
// ISA 7.12.2 16-bit layout: lane holds row/col (lane&15); per lane the 16 f16
// are K = k0 + 8*(lane>>4) + {0..7} and K = k0 + 16 + 8*(lane>>4) + {0..7}.
__device__ __forceinline__ v16h load_tile16(const _Float16* base, int row,
                                            int stride, int khalf, int k0) {
  const _Float16* p = base + row * stride + k0 + khalf * 8;
  v8h lo = *(const v8h*)(p);
  v8h hi = *(const v8h*)(p + 16);
  return __builtin_shufflevector(lo, hi, 0, 1, 2, 3, 4, 5, 6, 7,
                                 8, 9, 10, 11, 12, 13, 14, 15);
}

__global__ __launch_bounds__(NT) void senet_fused(
    const float* __restrict__ x, const float* __restrict__ W1,
    const float* __restrict__ W2, float* __restrict__ out) {
  __shared__ float part[ROWS * NT];                    // 16 KB partial sums
  __shared__ alignas(16) _Float16 W1h[NR * NF];        // 16 KB  W1 [64][128] f16
  __shared__ alignas(16) _Float16 W2h[NF * NR];        // 16 KB  W2 [128][64] f16
  __shared__ alignas(16) _Float16 xxh[ROWS * NF];      //  4 KB  field means f16
  __shared__ alignas(16) _Float16 hidh[ROWS * NR];     //  2 KB  relu hidden f16
  __shared__ float gates[ROWS * NF];                   //  8 KB  sigmoid gates

  const int t     = threadIdx.x;
  const int wave  = t >> 5;        // wave-uniform
  const int lane  = t & 31;
  const int mn    = lane & 15;     // row (A) / col (B) index
  const int khalf = lane >> 4;     // K-half select
  const int row0  = blockIdx.x * ROWS;

  // Stage weights as f16 (tiny; L2-resident across blocks).
  for (int i = t; i < NR * NF; i += NT) W1h[i] = (_Float16)W1[i];
  for (int i = t; i < NF * NR; i += NT) W2h[i] = (_Float16)W2[i];

  // ---- Phase A: per-(row, chunk) partial sums. Thread t sums its aligned
  // 16-float chunk; fixed combine order later => deterministic.
  for (int r = 0; r < ROWS; ++r) {
    const v4f* p = (const v4f*)(x + (size_t)(row0 + r) * DTOT + t * 16);
    v4f s4 = p[0] + p[1] + p[2] + p[3];
    part[r * NT + t] = s4.x + s4.y + s4.z + s4.w;
  }
  __syncthreads();

  // Reduce partials -> field means (f16). 2048 (row,field) pairs, 8 per thread.
  for (int j = 0; j < 8; ++j) {
    int p = t * 8 + j;
    int r = p >> 7, f = p & 127;
    float s, inv;
    if (f < 64) {
      s = part[r * NT + f];
      inv = 1.f / 16.f;
    } else if (f < 96) {
      int i0 = 64 + ((f - 64) << 1);
      s = part[r * NT + i0] + part[r * NT + i0 + 1];
      inv = 1.f / 32.f;
    } else {
      int i0 = 128 + ((f - 96) << 2);
      s = (part[r * NT + i0] + part[r * NT + i0 + 1]) +
          (part[r * NT + i0 + 2] + part[r * NT + i0 + 3]);
      inv = 1.f / 64.f;
    }
    xxh[p] = (_Float16)(s * inv);
  }
  __syncthreads();

  // ---- Phase B, layer 1: hidden[16x64] = relu(xx[16x128] @ W1^T).
  // Waves 0..3 each own one 16-wide N-tile; K=128 -> 4 WMMA steps.
  if (wave < 4) {
    v8f acc = {};
#pragma unroll
    for (int ks = 0; ks < 4; ++ks) {
      v16h A  = load_tile16(xxh, mn, NF, khalf, ks * 32);
      v16h Bm = load_tile16(W1h, wave * 16 + mn, NF, khalf, ks * 32);
      acc = __builtin_amdgcn_wmma_f32_16x16x32_f16(
          false, A, false, Bm, (short)0, acc, false, false);
    }
#pragma unroll
    for (int v = 0; v < 8; ++v) {  // D layout: VGPR v -> M = v + 8*khalf
      int m = v + (khalf ? 8 : 0);
      float h = acc[v] > 0.f ? acc[v] : 0.f;
      hidh[m * NR + wave * 16 + mn] = (_Float16)h;
    }
  }
  __syncthreads();

  // ---- Phase B, layer 2: gates[16x128] = sigmoid(hidden[16x64] @ W2^T).
  // All 8 waves, one 16-wide N-tile each; K=64 -> 2 WMMA steps.
  {
    v8f acc = {};
#pragma unroll
    for (int ks = 0; ks < 2; ++ks) {
      v16h A  = load_tile16(hidh, mn, NR, khalf, ks * 32);
      v16h Bm = load_tile16(W2h, wave * 16 + mn, NR, khalf, ks * 32);
      acc = __builtin_amdgcn_wmma_f32_16x16x32_f16(
          false, A, false, Bm, (short)0, acc, false, false);
    }
#pragma unroll
    for (int v = 0; v < 8; ++v) {
      int m = v + (khalf ? 8 : 0);
      gates[m * NF + wave * 16 + mn] = 1.0f / (1.0f + __expf(-acc[v]));
    }
  }
  __syncthreads();

  // ---- Phase C: re-read x (tile is L2-hot), scale, non-temporal store.
  const int f_t = field_of_chunk(t);
  for (int r = 0; r < ROWS; ++r) {
    float g = gates[r * NF + f_t];
    const v4f* p = (const v4f*)(x + (size_t)(row0 + r) * DTOT + t * 16);
    v4f* q = (v4f*)(out + (size_t)(row0 + r) * DTOT + t * 16);
#pragma unroll
    for (int i = 0; i < 4; ++i) {
      v4f v = p[i] * g;
      __builtin_nontemporal_store(v, q + i);
    }
  }
}

extern "C" void kernel_launch(void* const* d_in, const int* in_sizes, int n_in,
                              void* d_out, int out_size, void* d_ws,
                              size_t ws_size, hipStream_t stream) {
  const float* x  = (const float*)d_in[0];  // [16384, 4096] f32
  const float* W1 = (const float*)d_in[1];  // [64, 128] f32
  const float* W2 = (const float*)d_in[2];  // [128, 64] f32
  float* out = (float*)d_out;               // [16384, 4096] f32

  const int nrows = in_sizes[0] / DTOT;     // 16384
  dim3 grid(nrows / ROWS);                  // 1024 blocks
  dim3 block(NT);
  senet_fused<<<grid, block, 0, stream>>>(x, W1, W2, out);
}
